// DGCNN_23072564314771
// MI455X (gfx1250) — matrix operand order repaired and checked by reading, not compiled, not run on hardware
//
#include <hip/hip_runtime.h>
#include <math.h>

// ---------------------------------------------------------------------------
// DGCNN forward for MI455X (gfx1250, wave32, WMMA).
// B=4 clouds x P=2048 points, K=20 neighbors.
// - All GEMM K-dims are multiples of 32 (layer-1 input/weights zero-padded).
// - Weights pre-packed once into WMMA B-fragment bf16 layout:
//     packed[((tn*(K/32) + c32)*32 + lane)*16 + t]
//   so each B fragment is a single 32-byte vector load.
// - Generic GEMM blocks 4 M-tiles per wave (B fragment reused 4x).
// ---------------------------------------------------------------------------

#define NB   4
#define NP   2048
#define NPTS (NB * NP)
#define KNN  20

typedef __attribute__((ext_vector_type(16))) __bf16 v16bf;
typedef __attribute__((ext_vector_type(8)))  __bf16 v8bf;
typedef __attribute__((ext_vector_type(8)))  float  v8f;

// ---- WMMA fragment index maps (CDNA5 ISA 7.12.2, 16-bit A 16x32 / B 32x16) ----
// A: lane l (half = l>>4, mr = l&15) holds row M=mr;
//    element t<8  -> K = half*8 + t ; element t>=8 -> K = 16 + half*8 + (t-8)
// B: lane n holds col N = n&15; element t -> K same map as A
// C/D: VGPR r, lane l -> (M = r + 8*half, N = l&15)

__device__ __forceinline__ void cvt4(float4 v, v16bf& d, int o) {
    d[o + 0] = (__bf16)v.x; d[o + 1] = (__bf16)v.y;
    d[o + 2] = (__bf16)v.z; d[o + 3] = (__bf16)v.w;
}
__device__ __forceinline__ void split4(float4 v, v16bf& hi, v16bf& lo, int o) {
    float f0 = v.x, f1 = v.y, f2 = v.z, f3 = v.w;
    __bf16 h0 = (__bf16)f0, h1 = (__bf16)f1, h2 = (__bf16)f2, h3 = (__bf16)f3;
    hi[o + 0] = h0; hi[o + 1] = h1; hi[o + 2] = h2; hi[o + 3] = h3;
    lo[o + 0] = (__bf16)(f0 - (float)h0); lo[o + 1] = (__bf16)(f1 - (float)h1);
    lo[o + 2] = (__bf16)(f2 - (float)h2); lo[o + 3] = (__bf16)(f3 - (float)h3);
}

// ---------------------------------------------------------------------------
// 0a) padding kernels for layer 1 (make K a multiple of 32)
// ---------------------------------------------------------------------------
__global__ void pad_x_kernel(const float* __restrict__ x, float* __restrict__ xp) {
    int p = blockIdx.x * blockDim.x + threadIdx.x;
    if (p >= NPTS) return;
    float* r = xp + (size_t)p * 32;
    r[0] = x[p * 3 + 0]; r[1] = x[p * 3 + 1]; r[2] = x[p * 3 + 2];
#pragma unroll
    for (int c = 3; c < 32; ++c) r[c] = 0.f;
}
// wp[64][64]: rows 0..2 = w1a rows 0..2 (Wa1), rows 32..34 = w1a rows 3..5 (Wa2)
__global__ void pad_w1a_kernel(const float* __restrict__ wa, float* __restrict__ wp) {
    int t = blockIdx.x * blockDim.x + threadIdx.x;
    if (t >= 64 * 64) return;
    int r = t >> 6, c = t & 63;
    float v = 0.f;
    if (r < 3) v = wa[r * 64 + c];
    else if (r >= 32 && r < 35) v = wa[(r - 32 + 3) * 64 + c];
    wp[t] = v;
}

// ---------------------------------------------------------------------------
// 0b) pack W[K,N] (f32, row-major, rows [wRowOff, wRowOff+K)) into the WMMA
//     B-fragment bf16 layout: one 32B slot per (n-tile, k-chunk, lane).
// ---------------------------------------------------------------------------
__global__ void pack_w_kernel(const float* __restrict__ W, int ldW, int wRowOff,
                              int K, int N, __bf16* __restrict__ out) {
    int id = blockIdx.x * blockDim.x + threadIdx.x;
    int nkc = K >> 5;
    int total = (N >> 4) * nkc * 32;
    if (id >= total) return;
    int lane = id & 31;
    int rest = id >> 5;
    int tn = rest / nkc, c32 = rest % nkc;
    int half = lane >> 4, col = tn * 16 + (lane & 15);
    __bf16* dst = out + (size_t)id * 16;
#pragma unroll
    for (int t = 0; t < 16; ++t) {
        int k = c32 * 32 + (t < 8 ? half * 8 + t : 16 + half * 8 + (t - 8));
        dst[t] = (__bf16)W[(size_t)(wRowOff + k) * ldW + col];
    }
}

// ---------------------------------------------------------------------------
// 1) squared norms per point over the layer's input feature slice
// ---------------------------------------------------------------------------
__global__ void sqnorm_kernel(const float* __restrict__ X, int ldIn, int colOff,
                              int D, float* __restrict__ sq) {
    int p = blockIdx.x * blockDim.x + threadIdx.x;
    if (p >= NPTS) return;
    const float* r = X + (size_t)p * ldIn + colOff;
    float s = 0.f;
    for (int c = 0; c < D; ++c) { float v = r[c]; s += v * v; }
    sq[p] = s;
}

// ---------------------------------------------------------------------------
// 2) dist[b,i,j] = sq_i + sq_j - 2 * <x_i, x_j>, Gram via split-bf16 WMMA
//    (hi*hi + hi*lo + lo*hi ~ f32-accurate). One wave per 16x16 tile. D%32==0.
// ---------------------------------------------------------------------------
__global__ void gram_dist_kernel(const float* __restrict__ X, int ldIn, int colOff,
                                 int D, const float* __restrict__ sq,
                                 float* __restrict__ dist) {
    const int TN = NP / 16;  // 128
    int wid = blockIdx.x * (blockDim.x >> 5) + (threadIdx.x >> 5);
    int total = NB * TN * TN;
    if (wid >= total) return;                 // wave-uniform guard
    int b  = wid / (TN * TN);
    int r2 = wid % (TN * TN);
    int ti = r2 / TN, tj = r2 % TN;
    int lane = threadIdx.x & 31;
    int half = lane >> 4, mr = lane & 15;
    int rowI = b * NP + ti * 16 + mr;   // A row (global point id)
    int colJ = b * NP + tj * 16 + mr;   // B col (global point id)
    const float* arow = X + (size_t)rowI * ldIn + colOff;
    const float* brow = X + (size_t)colJ * ldIn + colOff;

    v8f acc = {};
    for (int kc = 0; kc < D; kc += 32) {
        v16bf ahi, alo, bhi, blo;
        const float* ap = arow + kc + half * 8;
        split4(*(const float4*)(ap +  0), ahi, alo, 0);
        split4(*(const float4*)(ap +  4), ahi, alo, 4);
        split4(*(const float4*)(ap + 16), ahi, alo, 8);
        split4(*(const float4*)(ap + 20), ahi, alo, 12);
        const float* bp = brow + kc + half * 8;
        split4(*(const float4*)(bp +  0), bhi, blo, 0);
        split4(*(const float4*)(bp +  4), bhi, blo, 4);
        split4(*(const float4*)(bp + 16), bhi, blo, 8);
        split4(*(const float4*)(bp + 20), bhi, blo, 12);
        acc = __builtin_amdgcn_wmma_f32_16x16x32_bf16(false, ahi, false, bhi, (short)0, acc, false, false);
        acc = __builtin_amdgcn_wmma_f32_16x16x32_bf16(false, ahi, false, blo, (short)0, acc, false, false);
        acc = __builtin_amdgcn_wmma_f32_16x16x32_bf16(false, alo, false, bhi, (short)0, acc, false, false);
    }
    float sj = sq[colJ];
#pragma unroll
    for (int r = 0; r < 8; ++r) {
        int m  = r + 8 * half;
        int gi = b * NP + ti * 16 + m;
        float d = sq[gi] + sj - 2.f * acc[r];
        dist[(size_t)gi * NP + tj * 16 + (lane & 15)] = d;
    }
}

// ---------------------------------------------------------------------------
// 3) top-K smallest per dist row (includes self, like reference top_k(-dist)).
//    One 256-thread workgroup per point; row in LDS; 20 argmin passes.
//    Stores GLOBAL neighbor indices.
// ---------------------------------------------------------------------------
__global__ void topk_kernel(const float* __restrict__ dist, int* __restrict__ idx) {
    __shared__ float sd[NP];
    __shared__ float rv[256];
    __shared__ int   ri[256];
    int p = blockIdx.x;
    int t = threadIdx.x;
    const float* row = dist + (size_t)p * NP;
    for (int j = t; j < NP; j += 256) sd[j] = row[j];
    __syncthreads();
    int gbase = (p / NP) * NP;
    for (int kk = 0; kk < KNN; ++kk) {
        float bv = 3.4e38f; int bi = 0;
        for (int j = t; j < NP; j += 256) {
            float v = sd[j];
            if (v < bv || (v == bv && j < bi)) { bv = v; bi = j; }
        }
        rv[t] = bv; ri[t] = bi;
        __syncthreads();
        for (int s = 128; s > 0; s >>= 1) {
            if (t < s) {
                float ov = rv[t + s]; int oi = ri[t + s];
                if (ov < rv[t] || (ov == rv[t] && oi < ri[t])) { rv[t] = ov; ri[t] = oi; }
            }
            __syncthreads();
        }
        if (t == 0) { idx[p * KNN + kk] = gbase + ri[0]; sd[ri[0]] = 3.4e38f; }
        __syncthreads();
    }
}

// ---------------------------------------------------------------------------
// 4) generic GEMM: Out[M,N] = act(In[:,colOff:+K] @ Wpacked + bias)
//    bf16 WMMA, f32 accumulate. 4 M-tiles (64 rows) per wave, packed-B
//    fragment reused 4x. K % 32 == 0, M % 64 == 0.
// ---------------------------------------------------------------------------
__global__ void gemm_kernel(const float* __restrict__ In, int ldIn, int inColOff,
                            const __bf16* __restrict__ Wp,
                            const float* __restrict__ bias,
                            float* __restrict__ Out, int ldOut, int outColOff,
                            int M, int N, int K, int doRelu) {
    int wid = blockIdx.x * (blockDim.x >> 5) + (threadIdx.x >> 5);
    int tilesN = N >> 4;
    int total  = (M >> 6) * tilesN;
    if (wid >= total) return;                 // wave-uniform guard
    int tmb = wid / tilesN, tn = wid % tilesN;
    int lane = threadIdx.x & 31;
    int half = lane >> 4, mr = lane & 15;
    int col = tn * 16 + mr;
    int nkc = K >> 5;
    const v16bf* wp = (const v16bf*)Wp + (size_t)tn * nkc * 32 + lane;
    const float* a0 = In + (size_t)(tmb * 64 + mr) * ldIn + inColOff + half * 8;

    v8f acc[4] = {{}, {}, {}, {}};
    for (int c32 = 0; c32 < nkc; ++c32) {
        v16bf bf = wp[(size_t)c32 * 32];      // 32B packed B fragment
        int kc = c32 * 32;
#pragma unroll
        for (int mb = 0; mb < 4; ++mb) {
            const float* ap = a0 + (size_t)(mb * 16) * ldIn + kc;
            if (mb == 0) __builtin_prefetch(ap + 32, 0, 3);
            v16bf a;
            cvt4(*(const float4*)(ap +  0), a, 0);
            cvt4(*(const float4*)(ap +  4), a, 4);
            cvt4(*(const float4*)(ap + 16), a, 8);
            cvt4(*(const float4*)(ap + 20), a, 12);
            acc[mb] = __builtin_amdgcn_wmma_f32_16x16x32_bf16(false, a, false, bf, (short)0, acc[mb], false, false);
        }
    }
    float bv = bias ? bias[col] : 0.f;
#pragma unroll
    for (int mb = 0; mb < 4; ++mb) {
#pragma unroll
        for (int r = 0; r < 8; ++r) {
            int m = tmb * 64 + mb * 16 + r + 8 * half;
            float v = acc[mb][r] + bv;
            if (doRelu) v = fmaxf(v, 0.f);
            Out[(size_t)m * ldOut + outColOff + col] = v;
        }
    }
}

// ---------------------------------------------------------------------------
// 5) edge MLP layer-b + max over K neighbors, one 256-thread block per point.
//    h1[k] = relu(U_i + V_{j_k} - V_i); out = max_{k<20} relu(h1 @ Wb + bb)
//    h1 staged in LDS as bf16 (rows padded to 32 = two WMMA M-tiles);
//    8 waves split the DOUT/16 column tiles; packed-B fragment reused 2x.
// ---------------------------------------------------------------------------
template <int DH, int DOUT>
__global__ void edge_kernel(const float* __restrict__ U, const float* __restrict__ V,
                            const int* __restrict__ idx,
                            const __bf16* __restrict__ Wbp, const float* __restrict__ bb,
                            float* __restrict__ Out, int ldOut, int outColOff) {
    __shared__ __attribute__((aligned(16))) __bf16 sh1[32 * DH];
    __shared__ float  sUi[DH];
    __shared__ float  sVi[DH];
    __shared__ int    sidx[32];
    int i = blockIdx.x;
    int t = threadIdx.x;
    if (t < 32) sidx[t] = (t < KNN) ? idx[i * KNN + t] : 0;
    for (int c = t; c < DH; c += 256) {
        sUi[c] = U[(size_t)i * DH + c];
        sVi[c] = V[(size_t)i * DH + c];
    }
    __syncthreads();
    for (int e = t; e < 32 * DH; e += 256) {
        int kk = e / DH, c = e % DH;
        float v = 0.f;
        if (kk < KNN) {
            int j = sidx[kk];
            v = fmaxf(sUi[c] + V[(size_t)j * DH + c] - sVi[c], 0.f);
        }
        sh1[kk * DH + c] = (__bf16)v;
    }
    __syncthreads();

    int w = t >> 5, lane = t & 31;
    int half = lane >> 4, mr = lane & 15;
    const int nkc = DH >> 5;
    for (int tile = w; tile < (DOUT >> 4); tile += 8) {   // wave-uniform loop
        int col = tile * 16 + mr;
        const v16bf* wp = (const v16bf*)Wbp + (size_t)tile * nkc * 32 + lane;
        v8f c0 = {}, c1 = {};
        for (int c32 = 0; c32 < nkc; ++c32) {
            int kc = c32 * 32;
            // A fragments: two aligned 16B runs per 16-row tile (ds_load_b128)
            const __bf16* p0 = &sh1[mr * DH + kc + half * 8];
            const __bf16* p1 = &sh1[(16 + mr) * DH + kc + half * 8];
            v8bf a0l = *(const v8bf*)(p0);
            v8bf a0h = *(const v8bf*)(p0 + 16);
            v8bf a1l = *(const v8bf*)(p1);
            v8bf a1h = *(const v8bf*)(p1 + 16);
            v16bf a0 = __builtin_shufflevector(a0l, a0h, 0,1,2,3,4,5,6,7,8,9,10,11,12,13,14,15);
            v16bf a1 = __builtin_shufflevector(a1l, a1h, 0,1,2,3,4,5,6,7,8,9,10,11,12,13,14,15);
            v16bf bf = wp[(size_t)c32 * 32];  // 32B packed B fragment
            c0 = __builtin_amdgcn_wmma_f32_16x16x32_bf16(false, a0, false, bf, (short)0, c0, false, false);
            c1 = __builtin_amdgcn_wmma_f32_16x16x32_bf16(false, a1, false, bf, (short)0, c1, false, false);
        }
        float bv = bb[col];
        float mx = 0.f;  // relu outputs >= 0; 20 real rows always present
#pragma unroll
        for (int r = 0; r < 8; ++r)                       // rows r + 8*half in 0..15
            mx = fmaxf(mx, fmaxf(c0[r] + bv, 0.f));
        if (half == 0) {                                  // rows 16..19 only
#pragma unroll
            for (int r = 0; r < 4; ++r)
                mx = fmaxf(mx, fmaxf(c1[r] + bv, 0.f));
        }
        mx = fmaxf(mx, __shfl_xor(mx, 16, 32));           // combine lane halves
        if (half == 0)
            Out[(size_t)i * ldOut + outColOff + col] = mx;
    }
}

// ---------------------------------------------------------------------------
// 6) final 256 -> 1 projection + sigmoid
// ---------------------------------------------------------------------------
__global__ void fc3_kernel(const float* __restrict__ H, const float* __restrict__ w,
                           const float* __restrict__ b, float* __restrict__ out) {
    int p = blockIdx.x * blockDim.x + threadIdx.x;
    if (p >= NPTS) return;
    float z = b[0];
    for (int c = 0; c < 256; ++c) z += H[(size_t)p * 256 + c] * w[c];
    out[p] = 1.f / (1.f + expf(-z));
}

// ---------------------------------------------------------------------------
extern "C" void kernel_launch(void* const* d_in, const int* in_sizes, int n_in,
                              void* d_out, int out_size, void* d_ws, size_t ws_size,
                              hipStream_t stream) {
    (void)in_sizes; (void)n_in; (void)out_size; (void)ws_size;
    const float* x   = (const float*)d_in[0];
    // d_in[1] is `batch` (int64): clouds are equal-size and sorted -> unused.
    const float* w1a = (const float*)d_in[2];  const float* b1a = (const float*)d_in[3];
    const float* w1b = (const float*)d_in[4];  const float* b1b = (const float*)d_in[5];
    const float* w2a = (const float*)d_in[6];  const float* b2a = (const float*)d_in[7];
    const float* w2b = (const float*)d_in[8];  const float* b2b = (const float*)d_in[9];
    const float* w3a = (const float*)d_in[10]; const float* b3a = (const float*)d_in[11];
    const float* w3b = (const float*)d_in[12]; const float* b3b = (const float*)d_in[13];
    const float* w4a = (const float*)d_in[14]; const float* b4a = (const float*)d_in[15];
    const float* w4b = (const float*)d_in[16]; const float* b4b = (const float*)d_in[17];
    const float* fw1 = (const float*)d_in[18]; const float* fb1 = (const float*)d_in[19];
    const float* fw2 = (const float*)d_in[20]; const float* fb2 = (const float*)d_in[21];
    const float* fw3 = (const float*)d_in[22]; const float* fb3 = (const float*)d_in[23];

    char* ws = (char*)d_ws;
    size_t off = 0;
    auto alloc = [&](size_t bytes) -> void* {
        void* p = ws + off;
        off += (bytes + 255) & ~(size_t)255;
        return p;
    };
    float* xcat = (float*)alloc((size_t)NPTS * 960 * sizeof(float)); // concat features
    float* sq   = (float*)alloc((size_t)NPTS * sizeof(float));
    float* dist = (float*)alloc((size_t)NPTS * NP * sizeof(float));
    int*   idx  = (int*)  alloc((size_t)NPTS * KNN * sizeof(int));
    float* U    = (float*)alloc((size_t)NPTS * 512 * sizeof(float));
    float* V    = (float*)alloc((size_t)NPTS * 512 * sizeof(float));
    float* h1   = (float*)alloc((size_t)NPTS * 512 * sizeof(float));
    float* h2   = (float*)alloc((size_t)NPTS * 256 * sizeof(float));
    float* xp   = (float*)alloc((size_t)NPTS * 32 * sizeof(float));  // padded layer-1 x
    float* wp   = (float*)alloc((size_t)64 * 64 * sizeof(float));    // padded w1a
    // packed bf16 weights (WMMA B-fragment layout)
    __bf16* p1aU = (__bf16*)alloc((size_t)32 * 64 * 2);
    __bf16* p1aV = (__bf16*)alloc((size_t)32 * 64 * 2);
    __bf16* p2aU = (__bf16*)alloc((size_t)64 * 128 * 2);
    __bf16* p2aV = (__bf16*)alloc((size_t)64 * 128 * 2);
    __bf16* p3aU = (__bf16*)alloc((size_t)128 * 256 * 2);
    __bf16* p3aV = (__bf16*)alloc((size_t)128 * 256 * 2);
    __bf16* p4aU = (__bf16*)alloc((size_t)256 * 512 * 2);
    __bf16* p4aV = (__bf16*)alloc((size_t)256 * 512 * 2);
    __bf16* p1b  = (__bf16*)alloc((size_t)64 * 64 * 2);
    __bf16* p2b  = (__bf16*)alloc((size_t)128 * 128 * 2);
    __bf16* p3b  = (__bf16*)alloc((size_t)256 * 256 * 2);
    __bf16* p4b  = (__bf16*)alloc((size_t)512 * 512 * 2);
    __bf16* pf1  = (__bf16*)alloc((size_t)960 * 512 * 2);
    __bf16* pf2  = (__bf16*)alloc((size_t)512 * 256 * 2);

    pad_x_kernel<<<(NPTS + 255) / 256, 256, 0, stream>>>(x, xp);
    pad_w1a_kernel<<<(64 * 64 + 255) / 256, 256, 0, stream>>>(w1a, wp);

    auto pack = [&](const float* W, int ldW, int wRowOff, int K, int N, __bf16* out) {
        int total = (N >> 4) * (K >> 5) * 32;
        pack_w_kernel<<<(total + 255) / 256, 256, 0, stream>>>(W, ldW, wRowOff, K, N, out);
    };
    pack(wp,  64,  0,   32,  64,  p1aU); pack(wp,  64,  32,  32,  64,  p1aV);
    pack(w2a, 128, 0,   64,  128, p2aU); pack(w2a, 128, 64,  64,  128, p2aV);
    pack(w3a, 256, 0,   128, 256, p3aU); pack(w3a, 256, 128, 128, 256, p3aV);
    pack(w4a, 512, 0,   256, 512, p4aU); pack(w4a, 512, 256, 256, 512, p4aV);
    pack(w1b, 64,  0,   64,  64,  p1b);
    pack(w2b, 128, 0,   128, 128, p2b);
    pack(w3b, 256, 0,   256, 256, p3b);
    pack(w4b, 512, 0,   512, 512, p4b);
    pack(fw1, 512, 0,   960, 512, pf1);
    pack(fw2, 256, 0,   512, 256, pf2);

    auto run_layer = [&](const float* Xin, int ldIn, int colOff, int Kdim,
                         const __bf16* paU, const __bf16* paV,
                         const float* ba, const __bf16* pb, const float* bbp,
                         int DH, int outColOff, int layer) {
        sqnorm_kernel<<<(NPTS + 255) / 256, 256, 0, stream>>>(Xin, ldIn, colOff, Kdim, sq);
        int gtiles = NB * (NP / 16) * (NP / 16);
        gram_dist_kernel<<<(gtiles + 3) / 4, 128, 0, stream>>>(Xin, ldIn, colOff, Kdim, sq, dist);
        topk_kernel<<<NPTS, 256, 0, stream>>>(dist, idx);
        int utiles = (NPTS / 64) * (DH / 16);
        // U = Xin @ Wa1 + ba   (no relu: relu applied after V_j - V_i add)
        gemm_kernel<<<(utiles + 3) / 4, 128, 0, stream>>>(Xin, ldIn, colOff, paU,
                                                          ba, U, DH, 0, NPTS, DH, Kdim, 0);
        // V = Xin @ Wa2
        gemm_kernel<<<(utiles + 3) / 4, 128, 0, stream>>>(Xin, ldIn, colOff, paV,
                                                          nullptr, V, DH, 0, NPTS, DH, Kdim, 0);
        switch (layer) {
            case 1: edge_kernel<64, 64>  <<<NPTS, 256, 0, stream>>>(U, V, idx, pb, bbp, xcat, 960, outColOff); break;
            case 2: edge_kernel<128, 128><<<NPTS, 256, 0, stream>>>(U, V, idx, pb, bbp, xcat, 960, outColOff); break;
            case 3: edge_kernel<256, 256><<<NPTS, 256, 0, stream>>>(U, V, idx, pb, bbp, xcat, 960, outColOff); break;
            default: edge_kernel<512, 512><<<NPTS, 256, 0, stream>>>(U, V, idx, pb, bbp, xcat, 960, outColOff); break;
        }
    };

    // EdgeConv stack; xcat columns: [0,64) L1 | [64,192) L2 | [192,448) L3 | [448,960) L4
    run_layer(xp,   32,  0,   32,  p1aU, p1aV, b1a, p1b, b1b,  64,   0, 1);
    run_layer(xcat, 960, 0,   64,  p2aU, p2aV, b2a, p2b, b2b, 128,  64, 2);
    run_layer(xcat, 960, 64,  128, p3aU, p3aV, b3a, p3b, b3b, 256, 192, 3);
    run_layer(xcat, 960, 192, 256, p4aU, p4aV, b4a, p4b, b4b, 512, 448, 4);

    // FC head
    {
        int tiles = (NPTS / 64) * (512 / 16);
        gemm_kernel<<<(tiles + 3) / 4, 128, 0, stream>>>(xcat, 960, 0, pf1,
                                                         fb1, h1, 512, 0, NPTS, 512, 960, 1);
    }
    {
        int tiles = (NPTS / 64) * (256 / 16);
        gemm_kernel<<<(tiles + 3) / 4, 128, 0, stream>>>(h1, 512, 0, pf2,
                                                         fb2, h2, 256, 0, NPTS, 256, 512, 1);
    }
    fc3_kernel<<<(NPTS + 255) / 256, 256, 0, stream>>>(h2, fw3, fb3, (float*)d_out);
}
